// SparseFeedForward_45037027065974
// MI455X (gfx1250) — matrix operand order repaired and checked by reading, not compiled, not run on hardware
//
#include <hip/hip_runtime.h>
#include <hip/hip_bf16.h>

// Problem constants (from reference): B=4, S=2048, H=1024, E=8, TOP_K=2
#define TOK   8192      // B*S tokens
#define HDIM  1024      // hidden
#define NEXP  8         // experts

typedef __attribute__((ext_vector_type(16))) __bf16 v16bf;
typedef __attribute__((ext_vector_type(8)))  __bf16 v8bf;
typedef __attribute__((ext_vector_type(8)))  float  v8f;

// ---------------------------------------------------------------------------
// Kernel 1: split fp32 -> bf16 hi/lo pair (hi = RNE(v), lo = RNE(v - hi))
// ---------------------------------------------------------------------------
__global__ void split_bf16_kernel(const float* __restrict__ src,
                                  __bf16* __restrict__ hi,
                                  __bf16* __restrict__ lo, int n) {
    int i = blockIdx.x * blockDim.x + threadIdx.x;
    if (i < n) {
        float v = src[i];
        __bf16 h = (__bf16)v;
        hi[i] = h;
        lo[i] = (__bf16)(v - (float)h);
    }
}

// ---------------------------------------------------------------------------
// Kernel 2: gating. One wave32 per token: logits -> softmax -> top2 ->
// renormalize -> dense [T,8] weight vector (zeros outside top2).
// softmax normalization cancels in the top-2 renorm, so we use exp directly.
// ---------------------------------------------------------------------------
__global__ void gate_kernel(const float* __restrict__ x,
                            const float* __restrict__ Wg,
                            const float* __restrict__ bg,
                            float* __restrict__ wsout) {
    int wave = threadIdx.x >> 5;
    int lane = threadIdx.x & 31;
    int t = blockIdx.x * (blockDim.x >> 5) + wave;
    if (t >= TOK) return;
    const float* xr = x + (size_t)t * HDIM;
    float acc[NEXP];
#pragma unroll
    for (int e = 0; e < NEXP; ++e) acc[e] = 0.f;
    for (int k = lane; k < HDIM; k += 32) {
        float xv = xr[k];
#pragma unroll
        for (int e = 0; e < NEXP; ++e) acc[e] += xv * Wg[e * HDIM + k];
    }
#pragma unroll
    for (int e = 0; e < NEXP; ++e) {
        acc[e] += __shfl_xor(acc[e], 16, 32);
        acc[e] += __shfl_xor(acc[e], 8, 32);
        acc[e] += __shfl_xor(acc[e], 4, 32);
        acc[e] += __shfl_xor(acc[e], 2, 32);
        acc[e] += __shfl_xor(acc[e], 1, 32);
    }
    if (lane == 0) {
        float g[NEXP];
        float m = -1e30f;
#pragma unroll
        for (int e = 0; e < NEXP; ++e) { g[e] = acc[e] + bg[e]; m = fmaxf(m, g[e]); }
#pragma unroll
        for (int e = 0; e < NEXP; ++e) g[e] = __expf(g[e] - m);
        int i0 = 0;
#pragma unroll
        for (int e = 1; e < NEXP; ++e) if (g[e] > g[i0]) i0 = e;
        int i1 = (i0 == 0) ? 1 : 0;
#pragma unroll
        for (int e = 0; e < NEXP; ++e) if (e != i0 && g[e] > g[i1]) i1 = e;
        float denom = g[i0] + g[i1];
#pragma unroll
        for (int e = 0; e < NEXP; ++e) {
            float w = (e == i0) ? g[i0] / denom : ((e == i1) ? g[i1] / denom : 0.f);
            wsout[(size_t)t * NEXP + e] = w;
        }
    }
}

// ---------------------------------------------------------------------------
// Kernel 3: fused bf16x3 WMMA GEMM (8192 tokens x 8192 cols x 1024 K) with
// expert-combine epilogue.
//   - One wave computes a 32-token x 16-h output tile = 2 M-subtiles x
//     128 GEMM columns (16 h * 8 e), K=1024.
//   - The 4 waves of a block share the SAME 128-column B panel (consecutive
//     token tiles) so their B loads hit WGP$ together -> ~4x less L2 traffic.
//   - bf16x3 split precision: hi*hi + hi*lo + lo*hi  (~fp32 accuracy).
// ---------------------------------------------------------------------------
__global__ __launch_bounds__(128)
void moe_wmma_kernel(const __bf16* __restrict__ xh, const __bf16* __restrict__ xl,
                     const __bf16* __restrict__ wh, const __bf16* __restrict__ wl,
                     const float* __restrict__ bexp, const float* __restrict__ wsg,
                     float* __restrict__ out) {
    const int K = HDIM;
    int lane = threadIdx.x & 31;
    int wave = threadIdx.x >> 5;
    // 256 M-tiles (32 tokens each) x 64 h-tiles; block = 4 consecutive M-tiles
    // sharing one h-tile.
    int htile  = blockIdx.x & 63;
    int mgroup = blockIdx.x >> 6;          // 64 groups
    int mtile  = mgroup * 4 + wave;        // 0..255
    int t0     = mtile * 32;
    int jbase  = htile * 128;              // 16 h * 8 e columns

    v8f zero = {};
    v8f acc[2][8];
#pragma unroll
    for (int mt = 0; mt < 2; ++mt)
#pragma unroll
        for (int i = 0; i < 8; ++i) acc[mt][i] = zero;

    union Frag { v16bf v; v8bf h[2]; };

    int m  = lane & 15;
    int ka = (lane < 16) ? 0 : 8;    // A-matrix K offset for this lane half
    int kb = (lane < 16) ? 0 : 8;    // B-matrix K offset for this lane half

    const __bf16* xhrow0 = xh + (size_t)(t0 + m) * K;
    const __bf16* xlrow0 = xl + (size_t)(t0 + m) * K;
    const __bf16* xhrow1 = xhrow0 + (size_t)16 * K;
    const __bf16* xlrow1 = xlrow0 + (size_t)16 * K;

    for (int k0 = 0; k0 < K; k0 += 32) {
        // A fragments: 16x32 bf16. lanes 0-15 hold K {0..7,16..23}, lanes
        // 16-31 hold K {8..15,24..31} of rows M = lane&15 (CDNA5 ISA layout).
        Frag ah[2], al[2];
        ah[0].h[0] = *(const v8bf*)(xhrow0 + k0 + ka);
        ah[0].h[1] = *(const v8bf*)(xhrow0 + k0 + 16 + ka);
        al[0].h[0] = *(const v8bf*)(xlrow0 + k0 + ka);
        al[0].h[1] = *(const v8bf*)(xlrow0 + k0 + 16 + ka);
        ah[1].h[0] = *(const v8bf*)(xhrow1 + k0 + ka);
        ah[1].h[1] = *(const v8bf*)(xhrow1 + k0 + 16 + ka);
        al[1].h[0] = *(const v8bf*)(xlrow1 + k0 + ka);
        al[1].h[1] = *(const v8bf*)(xlrow1 + k0 + 16 + ka);
        __builtin_prefetch(xhrow0 + k0 + 32, 0, 1);
        __builtin_prefetch(xlrow0 + k0 + 32, 0, 1);
        __builtin_prefetch(xhrow1 + k0 + 32, 0, 1);
        __builtin_prefetch(xlrow1 + k0 + 32, 0, 1);

#pragma unroll
        for (int nt = 0; nt < 8; ++nt) {
            // B fragment: 32x16 bf16. lanes 0-15 -> col N=lane, K {0..7,16..23};
            // lanes 16-31 -> col N=lane-16, K {8..15,24..31}. Columns are W rows
            // (K-contiguous) so each lane does two 16B loads.
            const __bf16* whrow = wh + (size_t)(jbase + nt * 16 + m) * K + k0 + kb;
            const __bf16* wlrow = wl + (size_t)(jbase + nt * 16 + m) * K + k0 + kb;
            Frag bh, bl;
            bh.h[0] = ((const v8bf*)whrow)[0];
        bh.h[1] = ((const v8bf*)whrow)[2];
            bl.h[0] = ((const v8bf*)wlrow)[0];
        bl.h[1] = ((const v8bf*)wlrow)[2];
#pragma unroll
            for (int mt = 0; mt < 2; ++mt) {
                acc[mt][nt] = __builtin_amdgcn_wmma_f32_16x16x32_bf16(
                    false, ah[mt].v, false, bh.v, (short)0, acc[mt][nt], false, false);
                acc[mt][nt] = __builtin_amdgcn_wmma_f32_16x16x32_bf16(
                    false, ah[mt].v, false, bl.v, (short)0, acc[mt][nt], false, false);
                acc[mt][nt] = __builtin_amdgcn_wmma_f32_16x16x32_bf16(
                    false, al[mt].v, false, bh.v, (short)0, acc[mt][nt], false, false);
            }
        }
    }

    // Epilogue: acc[mt][nt] is a 16x16 F32 D-tile. VGPR r holds row r
    // (lanes 0-15) and row r+8 (lanes 16-31); lane column c = lane&15 maps to
    // j = jbase + nt*16 + c, i.e. e = c&7, h = htile*16 + nt*2 + (c>>3).
    // Reduce over e = 8-lane groups via shfl_xor; lanes {0,8,16,24} store.
    int c = lane & 15;
    int e = c & 7;
    int hoff = c >> 3;
    int tadd = (lane < 16) ? 0 : 8;
#pragma unroll
    for (int mt = 0; mt < 2; ++mt) {
        int tb = t0 + mt * 16;
        float wv[8];
#pragma unroll
        for (int r = 0; r < 8; ++r)
            wv[r] = wsg[(size_t)(tb + r + tadd) * NEXP + e];
#pragma unroll
        for (int nt = 0; nt < 8; ++nt) {
            int j = jbase + nt * 16 + c;
            float bj = bexp[j];
            int h = htile * 16 + nt * 2 + hoff;
#pragma unroll
            for (int r = 0; r < 8; ++r) {
                int t = tb + r + tadd;
                float v = (acc[mt][nt][r] + bj) * wv[r];
                v += __shfl_xor(v, 1, 32);
                v += __shfl_xor(v, 2, 32);
                v += __shfl_xor(v, 4, 32);
                if ((lane & 7) == 0) out[(size_t)t * HDIM + h] = v;
            }
        }
    }
}

// ---------------------------------------------------------------------------
extern "C" void kernel_launch(void* const* d_in, const int* in_sizes, int n_in,
                              void* d_out, int out_size, void* d_ws, size_t ws_size,
                              hipStream_t stream) {
    const float* x         = (const float*)d_in[0];   // [4,2048,1024]
    const float* W_experts = (const float*)d_in[1];   // [8192,1024]
    const float* b_experts = (const float*)d_in[2];   // [8192]
    const float* W_gate    = (const float*)d_in[3];   // [8,1024]
    const float* b_gate    = (const float*)d_in[4];   // [8]
    float* out = (float*)d_out;

    const int NX = TOK * HDIM;     // 8,388,608 (x elems == W elems)
    char* wsb = (char*)d_ws;
    __bf16* xh = (__bf16*)(wsb);
    __bf16* xl = (__bf16*)(wsb + 1ull * 2 * NX);
    __bf16* wh = (__bf16*)(wsb + 2ull * 2 * NX);
    __bf16* wl = (__bf16*)(wsb + 3ull * 2 * NX);
    float*  wg = (float*)(wsb + 4ull * 2 * NX);   // dense top-2 weights [TOK,8]

    // 1) fp32 -> bf16 hi/lo splits
    split_bf16_kernel<<<NX / 256, 256, 0, stream>>>(x, xh, xl, NX);
    split_bf16_kernel<<<NX / 256, 256, 0, stream>>>(W_experts, wh, wl, NX);

    // 2) gating: 8 tokens (8 wave32) per 256-thread block
    gate_kernel<<<TOK / 8, 256, 0, stream>>>(x, W_gate, b_gate, wg);

    // 3) fused WMMA GEMM + expert combine:
    //    grid = 64 m-groups x 64 h-tiles; 4 waves (consecutive m-tiles,
    //    shared B panel) per 128-thread block.
    moe_wmma_kernel<<<64 * 64, 128, 0, stream>>>(
        xh, xl, wh, wl, b_experts, wg, out);
}
